// nngls_26757646254418
// MI455X (gfx1250) — matrix operand order repaired and checked by reading, not compiled, not run on hardware
//
#include <hip/hip_runtime.h>
#include <hip/hip_bf16.h>

// ---------------------------------------------------------------------------
// NNGLS: WMMA MLP (TDM-staged weights in LDS) + per-node 20x20 Cholesky GP
// Target: gfx1250 (CDNA5), wave32
// ---------------------------------------------------------------------------

typedef __attribute__((ext_vector_type(16))) _Float16 v16h;
typedef __attribute__((ext_vector_type(8)))  float    v8f;
typedef unsigned int u32x4 __attribute__((ext_vector_type(4)));
typedef int          i32x8 __attribute__((ext_vector_type(8)));
typedef int          i32x4 __attribute__((ext_vector_type(4)));

#define GP_M     20
#define GP_TRI   210          // 20*21/2 lower-triangle entries
#define GP_BLOCK 64
#define EPS_DIST 1e-12f

#define MLP_H       256
#define MLP_P       64
#define MLP_STRIDE  72        // halves per LDS column: 64 data + 8 TDM pad
                              // (36 DWORDs -> 16-B aligned, 2-way banked)

// --------------------------------------------------------------------------
// Kernel 1: convert w1 [P=64, H=256] f32 row-major -> w1t [H][P] f16
// (column-major so each WMMA B-fragment is 16 contiguous halves)
// --------------------------------------------------------------------------
__global__ void convert_w1_kernel(const float* __restrict__ w1,
                                  _Float16* __restrict__ w1t,
                                  int P, int H) {
  int idx = blockIdx.x * blockDim.x + threadIdx.x;
  if (idx < P * H) {
    int col = idx / P;   // H index
    int k   = idx % P;   // P index
    w1t[col * P + k] = (_Float16)w1[k * H + col];
  }
}

// --------------------------------------------------------------------------
// Kernel 2: o = relu(x@w1 + b1) @ w2 + b2   via v_wmma_f32_16x16x32_f16
//
// Block = 8 waves. w1t (32 KB) is staged to LDS once per block by the
// Tensor Data Mover with pad_enable (4 DWORDs pad per 32 DWORDs), giving
// each 64-half column a 72-half stride. Waves then grid-stride over 16-row
// tiles: A fragments from x (f32->f16), B fragments from LDS (ds_load_b128),
// two K=32 WMMAs per column tile, fused ReLU + w2 accumulation, cross-lane
// reduce -> o.
//
// A layout (16-bit A 16x32, contiguous K): lanes 0-15 row M=lr hold K 0..15;
// lanes 16-31 same rows hold K 16..31.
// B layout: lane lr holds column ct*16+lr; lanes 0-15 K 0..15, lanes 16-31
// K 16..31. C/D: VGPR v, lanes 0-15 -> (M=v), lanes 16-31 -> (M=v+8).
// --------------------------------------------------------------------------
__global__ void __launch_bounds__(256)
mlp_wmma_kernel(const float* __restrict__ x,
                const _Float16* __restrict__ w1t,
                const float* __restrict__ b1,
                const float* __restrict__ w2,
                const float* __restrict__ b2,
                float* __restrict__ o, int N) {
  __shared__ _Float16 lds_w1[MLP_H * MLP_STRIDE];   // 36 KB

  // ---- stage w1t -> LDS via Tensor Data Mover (wave 0 issues) ----
  if (threadIdx.x < 32) {
    const unsigned long long ga = (unsigned long long)(uintptr_t)w1t;
    // low 32 bits of the flat shared address == LDS byte offset
    const unsigned lds_base =
        (unsigned)(unsigned long long)(uintptr_t)&lds_w1[0];

    u32x4 g0;
    g0[0] = 1u;                                  // count=1 (valid user D#)
    g0[1] = lds_base;                            // lds_addr
    g0[2] = (unsigned)(ga & 0xFFFFFFFFull);      // global_addr[31:0]
    g0[3] = (unsigned)((ga >> 32) & 0x01FFFFFFull) | 0x80000000u; // +type=2

    i32x8 g1;
    // data_size=2B (code 1), pad_enable=1, pad_interval=32DW (code 4),
    // pad_amount=4DW (code 3), workgroup_mask=0
    g1[0] = (int)0x07110000;
    g1[1] = (int)0x40000000;   // tensor_dim0[15:0]=16384 in bits[63:48]
    g1[2] = (int)0x00010000;   // tensor_dim1[15:0]=1     in bits[95:80]
    g1[3] = (int)0x40000000;   // tile_dim0=16384         in bits[127:112]
    g1[4] = (int)0x00000001;   // tile_dim1=1, tile_dim2=0
    g1[5] = (int)16384;        // tensor_dim0_stride[31:0]
    g1[6] = 0;
    g1[7] = 0;

    i32x4 gz = {0, 0, 0, 0};
#if __clang_major__ >= 23
    i32x8 gz8 = {0, 0, 0, 0, 0, 0, 0, 0};
    __builtin_amdgcn_tensor_load_to_lds(g0, g1, gz, gz, gz8, 0);
#else
    __builtin_amdgcn_tensor_load_to_lds(g0, g1, gz, gz, 0);
#endif
    __builtin_amdgcn_s_wait_tensorcnt(0);
  }
  __syncthreads();

  const int lane = threadIdx.x & 31;
  const int wave = threadIdx.x >> 5;
  const int hs   = lane >> 4;          // half-wave select
  const int lr   = lane & 15;
  const int nwaves = gridDim.x * (blockDim.x >> 5);
  const int gw     = blockIdx.x * (blockDim.x >> 5) + wave;

  for (int tile = gw; tile * 16 < N; tile += nwaves) {
    const int rowbase = tile * 16;
    const float* xr = x + (size_t)(rowbase + lr) * MLP_P;

    // A fragments for K-blocks [0,32) and [32,64)
    v16h a0, a1;
    const int ka = hs * 16;            // 0 or 16
#pragma unroll
    for (int t = 0; t < 16; ++t) {
      a0[t]     = (_Float16)xr[ka + t];
      a1[t]     = (_Float16)xr[32 + ka + t];
    }

    float acc[8];
#pragma unroll
    for (int v = 0; v < 8; ++v) acc[v] = 0.f;

    for (int ct = 0; ct < 16; ++ct) {
      const int col = ct * 16 + lr;
      const _Float16* wc = &lds_w1[col * MLP_STRIDE];
      const int kb = hs * 16;          // 0 or 16
      v16h bb0, bb1;
#pragma unroll
      for (int t = 0; t < 16; ++t) {
        bb0[t] = wc[kb + t];           // K = kb+t
        bb1[t] = wc[32 + kb + t];      // K = 32+kb+t
      }
      v8f c;
      const float binit = b1[col];     // same column for all 8 row-VGPRs
#pragma unroll
      for (int v = 0; v < 8; ++v) c[v] = binit;

      c = __builtin_amdgcn_wmma_f32_16x16x32_f16(false, a0, false, bb0,
                                                 (short)0, c, false, false);
      c = __builtin_amdgcn_wmma_f32_16x16x32_f16(false, a1, false, bb1,
                                                 (short)0, c, false, false);

      const float w2c = w2[col];
#pragma unroll
      for (int v = 0; v < 8; ++v)
        acc[v] = fmaf(fmaxf(c[v], 0.f), w2c, acc[v]);
    }

    // reduce over the 16 lanes sharing the same 8 rows
#pragma unroll
    for (int m = 1; m < 16; m <<= 1) {
#pragma unroll
      for (int v = 0; v < 8; ++v) acc[v] += __shfl_xor(acc[v], m, 32);
    }
    if (lr == 0) {
      const float bias2 = b2[0];
#pragma unroll
      for (int v = 0; v < 8; ++v)
        o[rowbase + hs * 8 + v] = acc[v] + bias2;
    }
  }
}

// --------------------------------------------------------------------------
// Kernel 3: per-node NNGP solve + decorrelation. One node per lane; the
// 210-entry covariance lower triangle lives in LDS laid out [tri][tid].
// --------------------------------------------------------------------------
__device__ __forceinline__ int tri_idx(int r, int c) { return (r * (r + 1)) / 2 + c; }

__global__ void __launch_bounds__(GP_BLOCK)
gp_decor_kernel(const float* __restrict__ pos, const float* __restrict__ y,
                const int* __restrict__ nbr, const float* __restrict__ theta,
                const float* __restrict__ o,
                float* __restrict__ y_decor, float* __restrict__ o_decor, int N) {
  extern __shared__ float smem[];
  float* Ab = smem + threadIdx.x;                 // stride GP_BLOCK per entry
  const int i = blockIdx.x * GP_BLOCK + threadIdx.x;
  if (i >= N) return;

  const float sigma_sq = theta[0];
  const float phi      = theta[1];
  const float tau      = theta[2];
  const float tau_sq   = tau * sigma_sq;

  const float pix = pos[2 * i], piy = pos[2 * i + 1];

  int   id[GP_M];
  float px[GP_M], py[GP_M], rhs[GP_M];
#pragma unroll
  for (int j = 0; j < GP_M; ++j) {
    id[j] = nbr[(size_t)i * GP_M + j];
    px[j] = pos[2 * id[j]];
    py[j] = pos[2 * id[j] + 1];
  }
#pragma unroll
  for (int j = 0; j < GP_M; ++j) {
    float dx = pix - px[j], dy = piy - py[j];
    rhs[j] = sigma_sq * __expf(-phi * sqrtf(fmaf(dx, dx, fmaf(dy, dy, EPS_DIST))));
  }

  // build lower triangle of Cov(N(i),N(i)) + tau_sq*I into LDS
  {
    int t = 0;
#pragma unroll 1
    for (int r = 0; r < GP_M; ++r) {
      for (int c = 0; c <= r; ++c) {
        float dx = px[r] - px[c], dy = py[r] - py[c];
        float cv = sigma_sq * __expf(-phi * sqrtf(fmaf(dx, dx, fmaf(dy, dy, EPS_DIST))));
        if (r == c) cv += tau_sq;
        Ab[t * GP_BLOCK] = cv;
        ++t;
      }
    }
  }

  // in-place Cholesky: A = L L^T (lower triangle holds L)
#pragma unroll 1
  for (int c = 0; c < GP_M; ++c) {
    float d = Ab[tri_idx(c, c) * GP_BLOCK];
    for (int k = 0; k < c; ++k) {
      float l = Ab[tri_idx(c, k) * GP_BLOCK];
      d = fmaf(-l, l, d);
    }
    const float ld  = sqrtf(d);
    const float inv = 1.f / ld;
    Ab[tri_idx(c, c) * GP_BLOCK] = ld;
#pragma unroll 1
    for (int r = c + 1; r < GP_M; ++r) {
      float s = Ab[tri_idx(r, c) * GP_BLOCK];
      for (int k = 0; k < c; ++k)
        s = fmaf(-Ab[tri_idx(r, k) * GP_BLOCK], Ab[tri_idx(c, k) * GP_BLOCK], s);
      Ab[tri_idx(r, c) * GP_BLOCK] = s * inv;
    }
  }

  // forward solve L z = rhs
  float z[GP_M];
#pragma unroll 1
  for (int r = 0; r < GP_M; ++r) {
    float s = rhs[r];
    for (int k = 0; k < r; ++k)
      s = fmaf(-Ab[tri_idx(r, k) * GP_BLOCK], z[k], s);
    z[r] = s / Ab[tri_idx(r, r) * GP_BLOCK];
  }
  // backward solve L^T b = z
  float b[GP_M];
#pragma unroll 1
  for (int r = GP_M - 1; r >= 0; --r) {
    float s = z[r];
    for (int k = r + 1; k < GP_M; ++k)
      s = fmaf(-Ab[tri_idx(k, r) * GP_BLOCK], b[k], s);
    b[r] = s / Ab[tri_idx(r, r) * GP_BLOCK];
  }

  float f = sigma_sq + tau;
#pragma unroll
  for (int j = 0; j < GP_M; ++j) f = fmaf(-b[j], rhs[j], f);
  const float invs = rsqrtf(f);

  float ys = y[i], os = o[i];
#pragma unroll
  for (int j = 0; j < GP_M; ++j) {
    ys = fmaf(-y[id[j]], b[j], ys);
    os = fmaf(-o[id[j]], b[j], os);
  }
  y_decor[i] = ys * invs;
  o_decor[i] = os * invs;
}

// --------------------------------------------------------------------------
// Launch: inputs (x, pos, y, nbr_idx, theta, w1, b1, w2, b2)
// Output: [y_decor | o_decor | o], 3*N floats.
// --------------------------------------------------------------------------
extern "C" void kernel_launch(void* const* d_in, const int* in_sizes, int n_in,
                              void* d_out, int out_size, void* d_ws, size_t ws_size,
                              hipStream_t stream) {
  const float* x     = (const float*)d_in[0];
  const float* pos   = (const float*)d_in[1];
  const float* y     = (const float*)d_in[2];
  const int*   nbr   = (const int*)d_in[3];
  const float* theta = (const float*)d_in[4];
  const float* w1    = (const float*)d_in[5];
  const float* b1    = (const float*)d_in[6];
  const float* w2    = (const float*)d_in[7];
  const float* b2    = (const float*)d_in[8];

  const int N = in_sizes[2];           // 200000
  const int H = in_sizes[6];           // 256
  const int P = in_sizes[0] / N;       // 64

  float* y_decor = (float*)d_out;
  float* o_decor = y_decor + N;
  float* o       = y_decor + 2 * (size_t)N;

  _Float16* w1t = (_Float16*)d_ws;     // 32 KB f16 col-major staging of w1

  // 1) stage w1 -> f16 column-major in global scratch
  const int cvt_n = P * H;
  convert_w1_kernel<<<(cvt_n + 255) / 256, 256, 0, stream>>>(w1, w1t, P, H);

  // 2) WMMA MLP: persistent blocks, TDM-staged weights in LDS
  const int waves    = (N + 15) / 16;
  int blocks         = (waves + 7) / 8;
  if (blocks > 640) blocks = 640;
  mlp_wmma_kernel<<<blocks, 256, 0, stream>>>(x, w1t, b1, w2, b2, o, N);

  // 3) per-node GP solve + decorrelation
  const size_t shmem = (size_t)GP_TRI * GP_BLOCK * sizeof(float);  // 53.76 KB
  gp_decor_kernel<<<(N + GP_BLOCK - 1) / GP_BLOCK, GP_BLOCK, shmem, stream>>>(
      pos, y, nbr, theta, o, y_decor, o_decor, N);
}
